// LTFGW_9612136808630
// MI455X (gfx1250) — matrix operand order repaired and checked by reading, not compiled, not run on hardware
//
#include <hip/hip_runtime.h>
#include <hip/hip_bf16.h>

#define N_NODES 20000
#define N_FEAT  64
#define N_TPL   10
#define TPL_N   10
#define K_NEIGH 15
#define KLOC    16
#define N_EDGES (N_NODES * 16)

typedef __attribute__((ext_vector_type(16))) _Float16 v16h;
typedef __attribute__((ext_vector_type(8)))  float    v8f;
typedef __attribute__((ext_vector_type(2)))  float    v2f;

// ---------------------------------------------------------------------------
// Cross-lane helpers (wave32).  xor-1/2/4 stay in the VALU via DPP8,
// xor-8 via DPP16 row_ror:8 (within a 16-lane row, (i+8)%16 == i^8),
// xor-16 (cross-row) via ds_swizzle with an immediate pattern.
// ---------------------------------------------------------------------------
__host__ __device__ constexpr int DPP8(int a, int b, int c, int d,
                                       int e, int f, int g, int h) {
    return a | (b << 3) | (c << 6) | (d << 9) | (e << 12) | (f << 15) |
           (g << 18) | (h << 21);
}

__device__ __forceinline__ float dpp_xor1(float x) {
    return __int_as_float(__builtin_amdgcn_mov_dpp8(__float_as_int(x),
                                                    DPP8(1, 0, 3, 2, 5, 4, 7, 6)));
}
__device__ __forceinline__ float dpp_xor2(float x) {
    return __int_as_float(__builtin_amdgcn_mov_dpp8(__float_as_int(x),
                                                    DPP8(2, 3, 0, 1, 6, 7, 4, 5)));
}
__device__ __forceinline__ float dpp_xor4(float x) {
    return __int_as_float(__builtin_amdgcn_mov_dpp8(__float_as_int(x),
                                                    DPP8(4, 5, 6, 7, 0, 1, 2, 3)));
}
__device__ __forceinline__ float dpp_xor8(float x) {
    // DPP16 row_ror:8  (ctrl 0x128), all rows/banks, bound_ctrl=1
    return __int_as_float(__builtin_amdgcn_update_dpp(
        __float_as_int(x), __float_as_int(x), 0x128, 0xf, 0xf, true));
}
__device__ __forceinline__ float swz_xor16(float x) {
    // group-of-32 swizzle: and=0x1f, or=0, xor=16  -> imm 0x401f
    return __int_as_float(__builtin_amdgcn_ds_swizzle(__float_as_int(x), 0x401f));
}

__device__ __forceinline__ float rowmax16(float x) {
    x = fmaxf(x, dpp_xor1(x));
    x = fmaxf(x, dpp_xor2(x));
    x = fmaxf(x, dpp_xor4(x));
    x = fmaxf(x, dpp_xor8(x));
    return x;
}
__device__ __forceinline__ float rowsum16(float x) {
    x += dpp_xor1(x);
    x += dpp_xor2(x);
    x += dpp_xor4(x);
    x += dpp_xor8(x);
    return x;
}

// Per-wave LDS ordering: DS ops of one wave execute in order; we only need to
// stop the compiler from reordering around the staging tile.
__device__ __forceinline__ void wave_fence() {
    __builtin_amdgcn_wave_barrier();
    asm volatile("" ::: "memory");
}

// ---------------------------------------------------------------------------
// Preprocessing: deterministic CSR build + first-15-by-edge-index neighbors
// ---------------------------------------------------------------------------
__global__ void zero_k(unsigned* p, int n) {
    for (int i = blockIdx.x * blockDim.x + threadIdx.x; i < n;
         i += gridDim.x * blockDim.x)
        p[i] = 0u;
}

__global__ void deg_k(const int* __restrict__ eidx, int* __restrict__ deg, int E) {
    int e = blockIdx.x * blockDim.x + threadIdx.x;
    if (e < E) atomicAdd(&deg[eidx[e]], 1);
}

__global__ void scan_k(const int* __restrict__ deg, int* __restrict__ start, int n) {
    __shared__ int part[1024];
    int t = threadIdx.x;
    int chunk = (n + 1023) >> 10;
    int b = t * chunk;
    int s = 0;
    for (int i = 0; i < chunk; ++i) { int j = b + i; if (j < n) s += deg[j]; }
    part[t] = s;
    __syncthreads();
    for (int off = 1; off < 1024; off <<= 1) {
        int v = (t >= off) ? part[t - off] : 0;
        __syncthreads();
        part[t] += v;
        __syncthreads();
    }
    int run = (t == 0) ? 0 : part[t - 1];
    for (int i = 0; i < chunk; ++i) {
        int j = b + i;
        if (j < n) { start[j] = run; run += deg[j]; }
    }
}

__global__ void scatter_k(const int* __restrict__ eidx, const int* __restrict__ start,
                          int* __restrict__ cursor, int* __restrict__ csr_e,
                          int* __restrict__ csr_d, int E) {
    int e = blockIdx.x * blockDim.x + threadIdx.x;
    if (e >= E) return;
    int s = eidx[e];
    int pos = start[s] + atomicAdd(&cursor[s], 1);
    csr_e[pos] = e;
    csr_d[pos] = eidx[E + e];
}

__global__ void neigh_k(const int* __restrict__ deg, const int* __restrict__ start,
                        const int* __restrict__ csr_e, const int* __restrict__ csr_d,
                        int* __restrict__ neigh, int* __restrict__ cnt, int n) {
    int i = blockIdx.x * blockDim.x + threadIdx.x;
    if (i >= n) return;
    int d = deg[i], st = start[i];
    int dm = d < K_NEIGH ? d : K_NEIGH;
    int be[K_NEIGH], bd[K_NEIGH];
    int c = 0;
    for (int j = 0; j < d; ++j) {
        int e = csr_e[st + j], dst = csr_d[st + j];
        if (c < K_NEIGH) {
            int p = c;
            while (p > 0 && be[p - 1] > e) { be[p] = be[p - 1]; bd[p] = bd[p - 1]; --p; }
            be[p] = e; bd[p] = dst; ++c;
        } else if (e < be[K_NEIGH - 1]) {
            int p = K_NEIGH - 1;
            while (p > 0 && be[p - 1] > e) { be[p] = be[p - 1]; bd[p] = bd[p - 1]; --p; }
            be[p] = e; bd[p] = dst;
        }
    }
    atomicAdd(&cnt[i], 1);  // self row
    for (int j = 0; j < K_NEIGH; ++j) {
        int dv = (j < dm) ? bd[j] : 0;  // masked slots -> node 0 (matches reference)
        neigh[i * K_NEIGH + j] = dv;
        atomicAdd(&cnt[dv], 1);
    }
}

// stats layout: [0]=S1, [1]=eps, [2]=alpha, [3]=pad, [4..67]=V1[64]
__global__ void xstats_k(const float* __restrict__ x, const int* __restrict__ cnt,
                         float* __restrict__ stats, int n) {
    __shared__ float sred[64];
    int f = threadIdx.x;
    int per = (n + gridDim.x - 1) / gridDim.x;
    int b0 = blockIdx.x * per;
    int b1 = b0 + per; if (b1 > n) b1 = n;
    float vf = 0.f, s1 = 0.f;
    for (int i = b0; i < b1; ++i) {
        float c = (float)cnt[i];
        if (c != 0.f) {
            float xv = x[(size_t)i * N_FEAT + f];
            vf += c * xv;
            s1 += c * xv * xv;
        }
    }
    atomicAdd(&stats[4 + f], vf);
    sred[f] = s1;
    __syncthreads();
    for (int off = 32; off > 0; off >>= 1) {
        if (f < off) sred[f] += sred[f + off];
        __syncthreads();
    }
    if (f == 0) atomicAdd(&stats[0], sred[0]);
}

__global__ void tstats_k(const float* __restrict__ tplF, const float* __restrict__ alpha0,
                         float* __restrict__ stats) {
    __shared__ float sv[64];
    __shared__ float sred[64];
    int f = threadIdx.x;
    float v = 0.f, s = 0.f;
    for (int j = 0; j < N_TPL * TPL_N; ++j) {
        float y = tplF[(size_t)j * N_FEAT + f];
        v += y;
        s += y * y;
    }
    sv[f] = v; sred[f] = s;
    __syncthreads();
    for (int off = 32; off > 0; off >>= 1) {
        if (f < off) sred[f] += sred[f + off];
        __syncthreads();
    }
    if (f == 0) {
        const float nk = (float)N_NODES * (float)KLOC;
        float S2 = sred[0];
        float S1 = stats[0];
        float dot = 0.f;
        for (int i = 0; i < 64; ++i)
            dot += (stats[4 + i] / nk) * (sv[i] / 100.f);
        // mean(M) = E|x|^2 + E|y|^2 - 2 E[x].E[y]   (exact factorization)
        float meanM = S1 / nk + S2 / 100.f - 2.f * dot;
        stats[1] = 0.05f * meanM + 1e-6f;
        stats[2] = 1.f / (1.f + __expf(-alpha0[0]));  // sigmoid
    }
}

// ---------------------------------------------------------------------------
// Main FGW kernel: block = 1 node, 10 waves = 10 templates (padded 10 -> 16).
// C/D layout: (v, lane) <-> (row = v + 8*half, col = lane&15).
// ---------------------------------------------------------------------------
__global__ __launch_bounds__(320) void fgw_main(
    const float* __restrict__ x, const float* __restrict__ tplA,
    const float* __restrict__ tplF, const int* __restrict__ neigh,
    const int* __restrict__ deg, const float* __restrict__ stats,
    float* __restrict__ out) {
    __shared__ float tmpT[N_TPL][256];  // per-wave staging tile (row k, col m)
    __shared__ float shX[16][64];       // xloc rows, staged once per block
    __shared__ int sh_nb[16];
    __shared__ float sh_sqx[16], sh_p[16], sh_logp[16], sh_hC1[16];
    __shared__ unsigned sh_mask;

    const int node = blockIdx.x;
    const int tid = threadIdx.x;
    const int t = tid >> 5;  // wave id == template id
    const int lane = tid & 31;
    const int half = lane >> 4;
    const int ln = lane & 15;

    // ---- phase 0a: local-graph metadata (16 threads) ----
    if (tid < 16) {
        int d = deg[node];
        int dm = d < K_NEIGH ? d : K_NEIGH;
        int valid = (tid >= 1 && tid <= dm);
        int g = (tid == 0) ? node : (valid ? neigh[node * K_NEIGH + tid - 1] : 0);
        sh_nb[tid] = g;
        float msum = (float)(1 + dm);
        int on = (tid == 0) || valid;
        sh_p[tid] = on ? 1.0f / msum : 0.0f;
        sh_logp[tid] = on ? __logf(1.0f / msum) : -1e9f;
        // hC1[k] = sum_l C1[k,l]^2 p_l  (C1 is the 0/1 star adjacency)
        sh_hC1[tid] = (tid == 0) ? (float)dm / msum : (valid ? 1.0f / msum : 0.0f);
        float sq = 0.f;
        const float* xr = x + (size_t)g * N_FEAT;
        for (int f = 0; f < N_FEAT; ++f) { float vv = xr[f]; sq += vv * vv; }
        sh_sqx[tid] = sq;
        if (tid == 0) {
            unsigned mb = 0;
            for (int j = 1; j <= dm; ++j) mb |= (1u << j);
            sh_mask = mb;
        }
    }
    __syncthreads();
    // ---- phase 0b: stage the 16 xloc rows into LDS (256 threads, b128) ----
    if (tid < 256) {
        int row = tid >> 4, c4 = (tid & 15) * 4;
        const float4 vv = *(const float4*)(x + (size_t)sh_nb[row] * N_FEAT + c4);
        *(float4*)&shX[row][c4] = vv;
    }
    __syncthreads();

    const float eps = stats[1];
    const float alpha = stats[2];
    const float inv_eps = 1.0f / eps;
    const unsigned mb = sh_mask;

    // ---- xloc f16 A fragments from LDS (16x32 layout per fragment) ----
    v16h a0, a1;
    {
        const float* rp = &shX[ln][0];
        for (int i = 0; i < 8; ++i) {
            a0[i]     = (_Float16)rp[8 * half + i];
            a0[8 + i] = (_Float16)rp[16 + 8 * half + i];
            a1[i]     = (_Float16)rp[32 + 8 * half + i];
            a1[8 + i] = (_Float16)rp[48 + 8 * half + i];
        }
    }
    // ---- template-feature f16 B fragments (32x16: lane col=ln) + |y|^2 ----
    const float* tf = tplF + ((size_t)t * TPL_N + ln) * N_FEAT;
    v16h b0, b1;
    float sqy = 0.f;
    for (int i = 0; i < 16; ++i) {
        float v0 = (ln < TPL_N) ? tf[16 * (i >> 3) + 8 * half + (i & 7)] : 0.f;
        float v1 = (ln < TPL_N) ? tf[32 + 16 * (i >> 3) + 8 * half + (i & 7)] : 0.f;
        b0[i] = (_Float16)v0;
        b1[i] = (_Float16)v1;
        sqy += v0 * v0 + v1 * v1;
    }
    sqy += swz_xor16(sqy);  // halves hold complementary feature ranges

    // ---- feature cost: M = |x|^2 + |y|^2 - 2 xloc . TF^T (two f16 WMMAs) ----
    v8f acc;
    for (int v = 0; v < 8; ++v) acc[v] = 0.f;
    acc = __builtin_amdgcn_wmma_f32_16x16x32_f16(false, a0, false, b0, (short)0, acc, false, false);
    acc = __builtin_amdgcn_wmma_f32_16x16x32_f16(false, a1, false, b1, (short)0, acc, false, false);

    float hc2 = 0.f;
    if (ln < TPL_N) {
        const float* c2r = tplA + ((size_t)t * TPL_N + ln) * TPL_N;
        for (int r = 0; r < TPL_N; ++r) { float vv = c2r[r]; hc2 += vv * vv; }
        hc2 *= (1.0f / TPL_N);
    }
    v8f M_, cC, logp_v;
    float maskf[8];
    for (int v = 0; v < 8; ++v) {
        int r = v + 8 * half;
        M_[v] = sh_sqx[r] + sqy - 2.0f * acc[v];
        cC[v] = sh_hC1[r] + hc2;
        logp_v[v] = sh_logp[r];
        maskf[v] = (float)((mb >> r) & 1u);
    }

    // C2^T as f32 B fragments: B[m][r] = C2[r][m], held in registers.
    v2f c2b[4];
    for (int s = 0; s < 4; ++s)
        for (int g2 = 0; g2 < 2; ++g2) {
            int m = 4 * s + 2 * half + g2;
            c2b[s][g2] = (ln < TPL_N && m < TPL_N)
                             ? tplA[((size_t)t * TPL_N + ln) * TPL_N + m] : 0.f;
        }

    // ---- Sinkhorn state ----
    v8f Tp, fv;
    for (int v = 0; v < 8; ++v) {
        Tp[v] = sh_p[v + 8 * half] * ((ln < TPL_N) ? (1.0f / TPL_N) : 0.f);
        fv[v] = 0.f;
    }
    float gl = 0.f;
    const float logq = (ln < TPL_N) ? __logf(1.0f / TPL_N) : -1e9f;

    // gw_tens(T) = constC - 2 * (C1 @ T) @ C2^T.  C1 is a star: C1@T needs only
    // a masked column-sum (row 0) and a broadcast of T's row 0 (other rows);
    // the dense tmp @ C2^T runs on 4x f32 WMMA with per-wave LDS staging.
    auto gw_tens = [&](const v8f& T) -> v8f {
        float t0 = T[0];  // value of row (half?8:0) at this column
        float tp0m = half ? swz_xor16(t0) : t0;  // T[0][m] in all lanes
        float cs = 0.f;
        for (int v = 0; v < 8; ++v) cs += maskf[v] * T[v];
        cs += swz_xor16(cs);  // colsum_m = sum_l mask_l T[l][m]
        v8f tmpv;
        for (int v = 0; v < 8; ++v) tmpv[v] = maskf[v] * tp0m;
        tmpv[0] = (half == 0) ? cs : tmpv[0];  // row 0 of tmp
        wave_fence();
        for (int v = 0; v < 8; ++v) tmpT[t][(v + 8 * half) * 16 + ln] = tmpv[v];
        wave_fence();
        v8f g2acc;
        for (int v = 0; v < 8; ++v) g2acc[v] = 0.f;
        const float2* rowp = (const float2*)&tmpT[t][ln * 16];  // A row = ln
        for (int s = 0; s < 4; ++s) {
            float2 pr = rowp[2 * s + half];  // K = 4s + 2*half + {0,1}
            v2f a;
            a[0] = pr.x;
            a[1] = pr.y;
            g2acc = __builtin_amdgcn_wmma_f32_16x16x4_f32(false, a, false, c2b[s],
                                                          (short)0, g2acc, false, false);
        }
        v8f gw;
        for (int v = 0; v < 8; ++v) gw[v] = cC[v] - 2.0f * g2acc[v];
        return gw;
    };

    for (int outer = 0; outer < 3; ++outer) {
        v8f gw = gw_tens(Tp);
        v8f lk;
        for (int v = 0; v < 8; ++v) {
            float G = (1.f - alpha) * M_[v] + 2.f * alpha * gw[v];
            lk[v] = -G * inv_eps;
        }
        for (int it = 0; it < 5; ++it) {
            // f = eps*(logp - LSE_m(logK + g/eps)) : row LSE, all in VALU DPP
            float gle = gl * inv_eps;
            for (int v = 0; v < 8; ++v) {
                float tv = lk[v] + gle;
                float mx = rowmax16(tv);
                float e = rowsum16(__expf(tv - mx));
                fv[v] = eps * (logp_v[v] - (mx + __logf(e)));
            }
            // g = eps*(logq - LSE_k(logK + f/eps)) : in-lane over v + halves
            float cmx = -3.0e38f;
            float tc[8];
            for (int v = 0; v < 8; ++v) {
                tc[v] = lk[v] + fv[v] * inv_eps;
                cmx = fmaxf(cmx, tc[v]);
            }
            cmx = fmaxf(cmx, swz_xor16(cmx));
            float cs = 0.f;
            for (int v = 0; v < 8; ++v) cs += __expf(tc[v] - cmx);
            cs += swz_xor16(cs);
            gl = eps * (logq - (cmx + __logf(cs)));
        }
        float gle = gl * inv_eps;
        for (int v = 0; v < 8; ++v)
            Tp[v] = __expf(lk[v] + fv[v] * inv_eps + gle);
    }

    // dist = (1-a) <M,Tp> + a <gw_tens(Tp),Tp>
    v8f gwF = gw_tens(Tp);
    float part = 0.f;
    for (int v = 0; v < 8; ++v)
        part += ((1.f - alpha) * M_[v] + alpha * gwF[v]) * Tp[v];
    part = rowsum16(part);
    part += swz_xor16(part);
    if (lane == 0) out[(size_t)node * N_TPL + t] = part;
}

// ---------------------------------------------------------------------------
extern "C" void kernel_launch(void* const* d_in, const int* in_sizes, int n_in,
                              void* d_out, int out_size, void* d_ws, size_t ws_size,
                              hipStream_t stream) {
    (void)in_sizes; (void)n_in; (void)out_size; (void)ws_size;
    const float* x      = (const float*)d_in[0];
    const int*   eidx   = (const int*)d_in[1];
    const float* tplA   = (const float*)d_in[2];
    const float* tplF   = (const float*)d_in[3];
    const float* alpha0 = (const float*)d_in[4];
    float* out = (float*)d_out;

    // workspace layout (~4.1 MB total)
    int* deg    = (int*)d_ws;                // N
    int* cursor = deg + N_NODES;             // N
    int* cnt    = cursor + N_NODES;          // N
    float* stats = (float*)(cnt + N_NODES);  // 68 floats
    int* start  = (int*)(stats + 68);        // N
    int* csr_e  = start + N_NODES;           // E
    int* csr_d  = csr_e + N_EDGES;           // E
    int* neigh  = csr_d + N_EDGES;           // N*15

    int zn = 3 * N_NODES + 68;  // deg, cursor, cnt, stats are contiguous
    zero_k<<<(zn + 255) / 256, 256, 0, stream>>>((unsigned*)d_ws, zn);
    deg_k<<<(N_EDGES + 255) / 256, 256, 0, stream>>>(eidx, deg, N_EDGES);
    scan_k<<<1, 1024, 0, stream>>>(deg, start, N_NODES);
    scatter_k<<<(N_EDGES + 255) / 256, 256, 0, stream>>>(eidx, start, cursor, csr_e, csr_d, N_EDGES);
    neigh_k<<<(N_NODES + 255) / 256, 256, 0, stream>>>(deg, start, csr_e, csr_d, neigh, cnt, N_NODES);
    xstats_k<<<80, 64, 0, stream>>>(x, cnt, stats, N_NODES);
    tstats_k<<<1, 64, 0, stream>>>(tplF, alpha0, stats);
    fgw_main<<<N_NODES, 320, 0, stream>>>(x, tplA, tplF, neigh, deg, stats, out);
}
